// LTIRnn_16621523436176
// MI455X (gfx1250) — matrix operand order repaired and checked by reading, not compiled, not run on hardware
//
#include <hip/hip_runtime.h>
#include <math.h>

// ---------------------------------------------------------------------------
// LTI RNN scan for MI455X (gfx1250, wave32, WMMA)
//   NX=64, NU=32, NY=32, NW=64, B=256, T=2048
// Derived (device-precomputed) matrices, all stored column-major [N][K] bf16:
//   Mz  (64x64): z  = x @ Mz  + u @ Gz          (lam folded in)
//   Gz  (32x64)
//   Myx (64x32): y  = x @ Myx + u @ Myu + w @ Myw
//   Myu (32x32), Myw (64x32)
//   Mxx (64x64): x' = x @ Mxx + u @ Mxu + w @ Mxw   (Y^-1 folded in)
//   Mxu (32x64), Mxw (64x64)
// ---------------------------------------------------------------------------

typedef unsigned short u16t;
typedef unsigned int   u32t;
typedef __attribute__((ext_vector_type(16))) u16t   v16u;
typedef __attribute__((ext_vector_type(8)))  u16t   v8u;
typedef __attribute__((ext_vector_type(8)))  u32t   v8i;
typedef __attribute__((ext_vector_type(16))) __bf16 v16bf;
typedef __attribute__((ext_vector_type(8)))  float  v8f;
typedef __attribute__((ext_vector_type(4)))  float  v4f;

constexpr int NX = 64, NU = 32, NY = 32, NW = 64;
constexpr int Bb = 256, Tt = 2048;

// bf16 element offsets inside d_ws / LDS weight block
constexpr int OFF_MZ  = 0;      // 64*64
constexpr int OFF_GZ  = 4096;   // 64*32
constexpr int OFF_MYX = 6144;   // 32*64
constexpr int OFF_MYU = 8192;   // 32*32
constexpr int OFF_MYW = 9216;   // 32*64
constexpr int OFF_MXX = 11264;  // 64*64
constexpr int OFF_MXU = 15360;  // 64*32
constexpr int OFF_MXW = 17408;  // 64*64
constexpr int WTOT    = 21504;

// scalar RNE f32->bf16 (setup kernel only; scan uses v_cvt_pk_bf16_f32)
__device__ __forceinline__ u16t f2bf(float f) {
  u32t u = __builtin_bit_cast(u32t, f);
  u = (u + 0x7FFFu + ((u >> 16) & 1u)) >> 16;
  return (u16t)u;
}

// CDNA5 hardware transcendental tanh; trailing v_nop covers the ISA's
// "1 independent op after TRANS before result use" hazard rule.
__device__ __forceinline__ float tanh_hw(float x) {
  float r;
  asm("v_tanh_f32 %0, %1\n\tv_nop" : "=v"(r) : "v"(x));
  return r;
}

// Packed f32x2 -> bf16x2 conversion (dst[15:0]=cvt(a), dst[31:16]=cvt(b))
__device__ __forceinline__ u32t pkbf(float a, float b) {
  u32t r;
  asm("v_cvt_pk_bf16_f32 %0, %1, %2" : "=v"(r) : "v"(a), "v"(b));
  return r;
}

// ---------------------------------------------------------------------------
// Setup kernel: invert Y (Gauss-Jordan in LDS) and bake derived bf16 matrices
// ---------------------------------------------------------------------------
__global__ __launch_bounds__(256) void ltirnn_setup(
    const float* __restrict__ Y,   const float* __restrict__ lam,
    const float* __restrict__ WA,  const float* __restrict__ WB1,
    const float* __restrict__ WB2, const float* __restrict__ WC1,
    const float* __restrict__ WD11,const float* __restrict__ WD12,
    const float* __restrict__ WC2, const float* __restrict__ WD21,
    u16t* __restrict__ ws) {
  __shared__ float aug[64 * 128];   // [Y | I]
  __shared__ float colp[64];
  __shared__ float pivs;
  const int tid = threadIdx.x;

  for (int i = tid; i < 8192; i += 256) {
    int r = i >> 7, c = i & 127;
    aug[i] = (c < 64) ? Y[r * 64 + c] : ((c - 64) == r ? 1.0f : 0.0f);
  }
  __syncthreads();

  for (int p = 0; p < 64; ++p) {
    if (tid == 0) pivs = 1.0f / aug[p * 128 + p];
    __syncthreads();
    if (tid < 128) aug[p * 128 + tid] *= pivs;
    __syncthreads();
    if (tid < 64) colp[tid] = aug[tid * 128 + p];
    __syncthreads();
    for (int i = tid; i < 8192; i += 256) {
      int r = i >> 7, c = i & 127;
      if (r != p) aug[i] -= colp[r] * aug[p * 128 + c];
    }
    __syncthreads();
  }
  // Yinv[m][n] = aug[m*128 + 64 + n]

  for (int idx = tid; idx < 4096; idx += 256) {           // Mz
    int n = idx >> 6, k = idx & 63;
    ws[OFF_MZ + idx] = f2bf(WC2[n * 64 + k] / lam[n]);
  }
  for (int idx = tid; idx < 2048; idx += 256) {           // Gz
    int n = idx >> 5, k = idx & 31;
    ws[OFF_GZ + idx] = f2bf(WD21[n * 32 + k] / lam[n]);
  }
  for (int idx = tid; idx < 2048; idx += 256) {           // Myx
    int n = idx >> 6, k = idx & 63;
    ws[OFF_MYX + idx] = f2bf(WC1[n * 64 + k]);
  }
  for (int idx = tid; idx < 1024; idx += 256) {           // Myu
    int n = idx >> 5, k = idx & 31;
    ws[OFF_MYU + idx] = f2bf(WD11[n * 32 + k]);
  }
  for (int idx = tid; idx < 2048; idx += 256) {           // Myw
    int n = idx >> 6, k = idx & 63;
    ws[OFF_MYW + idx] = f2bf(WD12[n * 64 + k]);
  }
  for (int idx = tid; idx < 4096; idx += 256) {           // Mxx = WA^T Yinv
    int n = idx >> 6, k = idx & 63;
    float s = 0.f;
    for (int m = 0; m < 64; ++m) s += WA[m * 64 + k] * aug[m * 128 + 64 + n];
    ws[OFF_MXX + idx] = f2bf(s);
  }
  for (int idx = tid; idx < 2048; idx += 256) {           // Mxu = WB1^T Yinv
    int n = idx >> 5, k = idx & 31;
    float s = 0.f;
    for (int m = 0; m < 64; ++m) s += WB1[m * 32 + k] * aug[m * 128 + 64 + n];
    ws[OFF_MXU + idx] = f2bf(s);
  }
  for (int idx = tid; idx < 4096; idx += 256) {           // Mxw = WB2^T Yinv
    int n = idx >> 6, k = idx & 63;
    float s = 0.f;
    for (int m = 0; m < 64; ++m) s += WB2[m * 64 + k] * aug[m * 128 + 64 + n];
    ws[OFF_MXW + idx] = f2bf(s);
  }
}

// ---------------------------------------------------------------------------
// Fragment loaders (CDNA5 WMMA 16x16x32 bf16 layouts, wave32)
//   A (16xK): lane L -> row M=L%16;  half j -> K = kBase + (j/8)*16 + (L/16)*8 + j%8
//   B (Kx16): lane L -> col N=L%16;  half j -> K = kBase + (j/8)*16 + (L/16)*8 + j%8
//   C/D     : lane L -> col N=L%16;  reg  i -> row M = i + 8*(L/16)
// ---------------------------------------------------------------------------
__device__ __forceinline__ v16bf bfrag(const u16t* wl, int off, int nTile,
                                       int Krows, int kTile, int lane) {
  int c  = nTile * 16 + (lane & 15);
  int k0 = kTile * 32 + ((lane >> 4) << 3);
  const u16t* p = wl + off + c * Krows + k0;
  v8u lo = *(const v8u*)(p);
  v8u hh = *(const v8u*)(p + 16);
  v16u r;
#pragma unroll
  for (int j = 0; j < 8; ++j) { r[j] = lo[j]; r[8 + j] = hh[j]; }
  return __builtin_bit_cast(v16bf, r);
}

__device__ __forceinline__ v16bf afrag_lds(const u16t* buf, int kTile, int lane) {
  int M = lane & 15, hi = lane >> 4;
  const u16t* p = buf + M * 64 + kTile * 32 + hi * 8;
  v8u lo = *(const v8u*)(p);
  v8u hh = *(const v8u*)(p + 16);
  v16u r;
#pragma unroll
  for (int j = 0; j < 8; ++j) { r[j] = lo[j]; r[8 + j] = hh[j]; }
  return __builtin_bit_cast(v16bf, r);
}

// rowptr points at (row M, K = kTile*32) of an f32 row-major matrix.
// Pairs of halves (2d, 2d+1) of the fragment form dword d -> v_cvt_pk_bf16_f32.
__device__ __forceinline__ v16bf afrag_global(const float* rowptr, int hi) {
  const float* p = rowptr + hi * 8;
  v4f a0 = *(const v4f*)(p);
  v4f a1 = *(const v4f*)(p + 4);
  v4f b0 = *(const v4f*)(p + 16);
  v4f b1 = *(const v4f*)(p + 20);
  v8i r;
  r[0] = pkbf(a0[0], a0[1]);
  r[1] = pkbf(a0[2], a0[3]);
  r[2] = pkbf(a1[0], a1[1]);
  r[3] = pkbf(a1[2], a1[3]);
  r[4] = pkbf(b0[0], b0[1]);
  r[5] = pkbf(b0[2], b0[3]);
  r[6] = pkbf(b1[0], b1[1]);
  r[7] = pkbf(b1[2], b1[3]);
  return __builtin_bit_cast(v16bf, r);
}

#define WMMA_BF16(A, Bf, C) \
  __builtin_amdgcn_wmma_f32_16x16x32_bf16(false, (A), false, (Bf), (short)0, (C), false, false)

// ---------------------------------------------------------------------------
// Scan kernel: 16 blocks x 1 wave32; each wave owns 16 batch rows for all T.
// ---------------------------------------------------------------------------
__global__ __launch_bounds__(32) void ltirnn_scan(
    const float* __restrict__ u, const float* __restrict__ h0,
    const u16t* __restrict__ wsrc, float* __restrict__ yout) {
  __shared__ alignas(16) u16t wl[WTOT];
  __shared__ alignas(16) u16t xb[16 * 64];
  __shared__ alignas(16) u16t wb[16 * 64];

  const int lane = threadIdx.x;
  {  // stage derived weights into LDS once
    const u32t* s = (const u32t*)wsrc;
    u32t* d = (u32t*)wl;
    for (int i = lane; i < WTOT / 2; i += 32) d[i] = s[i];
  }
  __syncthreads();

  const int col = lane & 15, hi = lane >> 4;
  const int rowBase = blockIdx.x * 16;
  const size_t uRowStride = (size_t)Tt * NU;

  // x0 = h0[1] -> A fragments (f32 -> bf16)
  const float* xrp = h0 + (size_t)Bb * NX + (size_t)(rowBase + col) * NX;
  v16bf xa0 = afrag_global(xrp, hi);
  v16bf xa1 = afrag_global(xrp + 32, hi);

  for (int t = 0; t < Tt; ++t) {
    const float* ub = u + (size_t)t * NU;
    const float* urp = ub + (size_t)(rowBase + col) * uRowStride;
    v16bf ua = afrag_global(urp, hi);
    if (t + 1 < Tt) __builtin_prefetch(urp + NU, 0, 1);  // global_prefetch next step

    // ---- z = x@Mz + u@Gz ; w = tanh(z) -> LDS (bf16 row-major 16x64)
#pragma unroll
    for (int n = 0; n < 4; ++n) {
      v8f z = {0, 0, 0, 0, 0, 0, 0, 0};
      z = WMMA_BF16(ua,  bfrag(wl, OFF_GZ, n, NU, 0, lane), z);
      z = WMMA_BF16(xa0, bfrag(wl, OFF_MZ, n, NX, 0, lane), z);
      z = WMMA_BF16(xa1, bfrag(wl, OFF_MZ, n, NX, 1, lane), z);
#pragma unroll
      for (int i = 0; i < 8; i += 2) {
        u32t pk = pkbf(tanh_hw(z[i]), tanh_hw(z[i + 1]));
        wb[(i + 8 * hi) * 64 + n * 16 + col]     = (u16t)pk;         // ds_store_b16
        wb[(i + 1 + 8 * hi) * 64 + n * 16 + col] = (u16t)(pk >> 16); // ds_store_b16_d16_hi
      }
    }
    __syncthreads();
    v16bf wa0 = afrag_lds(wb, 0, lane);
    v16bf wa1 = afrag_lds(wb, 1, lane);

    // ---- y = x@Myx + u@Myu + w@Myw -> global
#pragma unroll
    for (int n = 0; n < 2; ++n) {
      v8f y = {0, 0, 0, 0, 0, 0, 0, 0};
      y = WMMA_BF16(ua,  bfrag(wl, OFF_MYU, n, NU, 0, lane), y);
      y = WMMA_BF16(xa0, bfrag(wl, OFF_MYX, n, NX, 0, lane), y);
      y = WMMA_BF16(xa1, bfrag(wl, OFF_MYX, n, NX, 1, lane), y);
      y = WMMA_BF16(wa0, bfrag(wl, OFF_MYW, n, NX, 0, lane), y);
      y = WMMA_BF16(wa1, bfrag(wl, OFF_MYW, n, NX, 1, lane), y);
#pragma unroll
      for (int i = 0; i < 8; ++i)
        yout[((size_t)(rowBase + i + 8 * hi) * Tt + t) * NY + n * 16 + col] = y[i];
    }

    // ---- x' = x@Mxx + u@Mxu + w@Mxw -> LDS, reload as A fragments
#pragma unroll
    for (int n = 0; n < 4; ++n) {
      v8f xn = {0, 0, 0, 0, 0, 0, 0, 0};
      xn = WMMA_BF16(ua,  bfrag(wl, OFF_MXU, n, NU, 0, lane), xn);
      xn = WMMA_BF16(xa0, bfrag(wl, OFF_MXX, n, NX, 0, lane), xn);
      xn = WMMA_BF16(xa1, bfrag(wl, OFF_MXX, n, NX, 1, lane), xn);
      xn = WMMA_BF16(wa0, bfrag(wl, OFF_MXW, n, NX, 0, lane), xn);
      xn = WMMA_BF16(wa1, bfrag(wl, OFF_MXW, n, NX, 1, lane), xn);
#pragma unroll
      for (int i = 0; i < 8; i += 2) {
        u32t pk = pkbf(xn[i], xn[i + 1]);
        xb[(i + 8 * hi) * 64 + n * 16 + col]     = (u16t)pk;
        xb[(i + 1 + 8 * hi) * 64 + n * 16 + col] = (u16t)(pk >> 16);
      }
    }
    __syncthreads();
    xa0 = afrag_lds(xb, 0, lane);
    xa1 = afrag_lds(xb, 1, lane);
  }
}

// ---------------------------------------------------------------------------
extern "C" void kernel_launch(void* const* d_in, const int* in_sizes, int n_in,
                              void* d_out, int out_size, void* d_ws, size_t ws_size,
                              hipStream_t stream) {
  (void)in_sizes; (void)n_in; (void)out_size; (void)ws_size;
  const float* u    = (const float*)d_in[0];
  const float* h0   = (const float*)d_in[1];
  const float* Y    = (const float*)d_in[2];
  const float* lam  = (const float*)d_in[3];
  const float* WA   = (const float*)d_in[4];
  const float* WB1  = (const float*)d_in[5];
  const float* WB2  = (const float*)d_in[6];
  const float* WC1  = (const float*)d_in[7];
  const float* WD11 = (const float*)d_in[8];
  const float* WD12 = (const float*)d_in[9];
  const float* WC2  = (const float*)d_in[10];
  const float* WD21 = (const float*)d_in[11];
  u16t* ws = (u16t*)d_ws;

  ltirnn_setup<<<1, 256, 0, stream>>>(Y, lam, WA, WB1, WB2, WC1, WD11, WD12,
                                      WC2, WD21, ws);
  ltirnn_scan<<<16, 32, 0, stream>>>(u, h0, ws, (float*)d_out);
}
